// Net_Actor_44890998178498
// MI455X (gfx1250) — compile-verified, matrix operand order checked by reading, not run on hardware
//
#include <hip/hip_runtime.h>
#include <hip/hip_bf16.h>

// ---------------------------------------------------------------------------
// CDNA5 (gfx1250) implementation: GRU cell + 2x GAT layers + scoring head.
// GEMMs via v_wmma_f32_16x16x32_f16 (wave32 WMMA), weight staging via async
// global->LDS when available, edge aggregation via L2-resident float atomics.
// ---------------------------------------------------------------------------

typedef __attribute__((ext_vector_type(16))) _Float16 v16h;
typedef __attribute__((ext_vector_type(8)))  float    v8f;

union HFrag { uint4 u[2]; v16h v; };

#if __has_builtin(__builtin_amdgcn_global_load_async_to_lds_b128) && \
    __has_builtin(__builtin_amdgcn_s_wait_asynccnt)
#define USE_ASYNC_LDS 1
typedef int b128vec __attribute__((__vector_size__(16)));
__device__ __forceinline__ void async_g2l_b128(const void* g, void* l) {
    __builtin_amdgcn_global_load_async_to_lds_b128(
        (__attribute__((address_space(1))) b128vec*)g,
        (__attribute__((address_space(3))) b128vec*)l, 0, 0);
}
#endif

__device__ __forceinline__ float lrelu(float v) { return v > 0.f ? v : 0.2f * v; }

// order-preserving int encoding of float for atomicMax
__device__ __forceinline__ int enc_ord(float f) {
    int i = __float_as_int(f);
    return i >= 0 ? i : (i ^ 0x7fffffff);
}
__device__ __forceinline__ float dec_ord(int i) {
    return __int_as_float(i >= 0 ? i : (i ^ 0x7fffffff));
}
#define ENC_NEG_INF ((int)0x807FFFFF)   // enc_ord(-inf)

// ---------------------------------------------------------------------------
// f32 -> f16 conversion (grid-stride)
// ---------------------------------------------------------------------------
__global__ __launch_bounds__(256) void cvt_f32_f16(const float* __restrict__ src,
                                                   _Float16* __restrict__ dst, int n) {
    for (int i = blockIdx.x * blockDim.x + threadIdx.x; i < n;
         i += gridDim.x * blockDim.x)
        dst[i] = (_Float16)src[i];
}

// ---------------------------------------------------------------------------
// WMMA GEMM:  C[M x 128] = A[M x 128] * W^T (+bias),  W is 128 rows x 128 cols
// with row stride ldw (f16).  Optional f16 copy of C.
// Block: 256 threads = 8 waves, each wave owns a 16-row stripe, all 128 cols.
// ---------------------------------------------------------------------------
__global__ __launch_bounds__(256) void gemm_wmma_f16(
    const _Float16* __restrict__ A,   // M x 128, row-major f16
    const _Float16* __restrict__ W,   // 128 rows x 128 cols, row stride ldw
    int ldw,
    const float* __restrict__ bias,   // 128 or nullptr
    float* __restrict__ C,            // M x 128 f32
    _Float16* __restrict__ Ch,        // optional M x 128 f16 (or nullptr)
    int M)
{
    __shared__ __align__(16) _Float16 sW[128 * 136];   // padded rows: no bank conflicts

    const int tid = threadIdx.x;
    // cooperative staging of weights into LDS: 2048 16-byte chunks
#ifdef USE_ASYNC_LDS
    for (int c = tid; c < 2048; c += 256) {
        int row  = c >> 4;
        int col8 = (c & 15) << 3;
        async_g2l_b128(&W[row * ldw + col8], &sW[row * 136 + col8]);
    }
    __builtin_amdgcn_s_wait_asynccnt(0);
#else
    for (int c = tid; c < 2048; c += 256) {
        int row  = c >> 4;
        int col8 = (c & 15) << 3;
        *(uint4*)(&sW[row * 136 + col8]) = *(const uint4*)(&W[row * ldw + col8]);
    }
#endif
    __syncthreads();

    const int wave = tid >> 5;
    const int lane = tid & 31;
    const int row0 = blockIdx.x * 128 + wave * 16;
    if (row0 >= M) return;
    const int l15  = lane & 15;
    const int hi   = lane >> 4;          // 0: K {0..7,16..23}, 1: K {8..15,24..31}
    const long lrow = (long)(row0 + l15) * 128;

    v8f acc[8] = {};

#pragma unroll
    for (int kt = 0; kt < 4; ++kt) {
        const int k0 = kt * 32;
        HFrag af;
        af.u[0] = *(const uint4*)(&A[lrow + k0 + hi * 8]);
        af.u[1] = *(const uint4*)(&A[lrow + k0 + 16 + hi * 8]);
        // preload all 8 B fragments for this K-step so the WMMAs can issue
        // back-to-back instead of stalling on each pair of ds_loads
        HFrag bf[8];
#pragma unroll
        for (int nt = 0; nt < 8; ++nt) {
            const int nrow = nt * 16 + l15;          // output-dim row of W (= row of B^T)
            bf[nt].u[0] = *(const uint4*)(&sW[nrow * 136 + k0 + hi * 8]);
            bf[nt].u[1] = *(const uint4*)(&sW[nrow * 136 + k0 + 16 + hi * 8]);
        }
#pragma unroll
        for (int nt = 0; nt < 8; ++nt) {
            acc[nt] = __builtin_amdgcn_wmma_f32_16x16x32_f16(
                false, af.v, false, bf[nt].v, (short)0, acc[nt], false, false);
        }
    }

#pragma unroll
    for (int nt = 0; nt < 8; ++nt) {
        const int col = nt * 16 + l15;
        const float bv = bias ? bias[col] : 0.f;
#pragma unroll
        for (int r = 0; r < 8; ++r) {
            const long row = row0 + r + hi * 8;
            const float v = acc[nt][r] + bv;
            C[row * 128 + col] = v;
            if (Ch) Ch[row * 128 + col] = (_Float16)v;
        }
    }
}

// ---------------------------------------------------------------------------
// GRU branch (tiny: B=64 rows)
// ---------------------------------------------------------------------------
__global__ __launch_bounds__(128) void gru_inp_kernel(const float* __restrict__ input_,
                                                      const float* __restrict__ W_in,
                                                      float* __restrict__ inp, int L) {
    const int b = blockIdx.x, d = threadIdx.x;
    __shared__ float xbar[128];
    const float* base = input_ + (long)b * L * 128;
    float s = 0.f;
    for (int l = 0; l < L; ++l) s += base[l * 128 + d];
    xbar[d] = s / (float)L;
    __syncthreads();
    const float* wr = W_in + d * 128;
    float acc = 0.f;
    for (int k = 0; k < 128; ++k) acc += xbar[k] * wr[k];
    inp[b * 128 + d] = acc;
}

__global__ __launch_bounds__(128) void gru_gates_kernel(const float* __restrict__ state_,
                                                        const float* __restrict__ inp,
                                                        const float* __restrict__ W_z,
                                                        const float* __restrict__ W_r,
                                                        const float* __restrict__ W_h,
                                                        float* __restrict__ h_out) {
    const int b = blockIdx.x, d = threadIdx.x;
    __shared__ float st[128], ip[128], rs[128];
    st[d] = state_[b * 128 + d];
    ip[d] = inp[b * 128 + d];
    __syncthreads();
    const float* wz = W_z + d * 256;
    const float* wr = W_r + d * 256;
    float za = 0.f, ra = 0.f;
    for (int k = 0; k < 128; ++k) {
        za += st[k] * wz[k] + ip[k] * wz[128 + k];
        ra += st[k] * wr[k] + ip[k] * wr[128 + k];
    }
    const float z = 1.f / (1.f + __expf(-za));
    const float r = 1.f / (1.f + __expf(-ra));
    rs[d] = r * st[d];
    __syncthreads();
    const float* wh = W_h + d * 256;
    float ha = 0.f;
    for (int k = 0; k < 128; ++k) ha += rs[k] * wh[k] + ip[k] * wh[128 + k];
    const float hv = tanhf(ha);
    h_out[b * 128 + d] = (1.f - z) * st[d] + z * hv;
}

// ---------------------------------------------------------------------------
// GAT edge aggregation (wave per edge, 4 features per lane)
// ---------------------------------------------------------------------------
__global__ __launch_bounds__(256) void seg_init_kernel(int* __restrict__ m_enc,
                                                       float* __restrict__ denom,
                                                       float* __restrict__ num, long n) {
    for (long i = (long)blockIdx.x * blockDim.x + threadIdx.x; i < n;
         i += (long)gridDim.x * blockDim.x) {
        m_enc[i] = ENC_NEG_INF;
        denom[i] = 0.f;
        num[i]   = 0.f;
    }
}

__global__ __launch_bounds__(256) void edge_max_kernel(
    const int* __restrict__ src, const int* __restrict__ dst, int Emain, int total,
    const float* __restrict__ ai, const float* __restrict__ aj,
    const float* __restrict__ battn, int* __restrict__ m_enc)
{
    const int e = (blockIdx.x * 256 + threadIdx.x) >> 5;
    if (e >= total) return;
    const int lane = threadIdx.x & 31;
    int s, d;
    if (e < Emain) { s = src[e]; d = dst[e]; } else { s = d = e - Emain; }
    const long od = (long)d * 128 + lane * 4;
    const long os = (long)s * 128 + lane * 4;
    const float4 a  = *(const float4*)(&ai[od]);
    const float4 bj = *(const float4*)(&aj[os]);
    const float4 bb = *(const float4*)(&battn[lane * 4]);
    int* mp = m_enc + od;
    atomicMax(mp + 0, enc_ord(lrelu(a.x + bj.x + bb.x)));
    atomicMax(mp + 1, enc_ord(lrelu(a.y + bj.y + bb.y)));
    atomicMax(mp + 2, enc_ord(lrelu(a.z + bj.z + bb.z)));
    atomicMax(mp + 3, enc_ord(lrelu(a.w + bj.w + bb.w)));
}

__global__ __launch_bounds__(256) void edge_sum_kernel(
    const int* __restrict__ src, const int* __restrict__ dst, int Emain, int total,
    const float* __restrict__ ai, const float* __restrict__ aj,
    const float* __restrict__ battn, const float* __restrict__ xlin,
    const int* __restrict__ m_enc, float* __restrict__ denom, float* __restrict__ num)
{
    const int e = (blockIdx.x * 256 + threadIdx.x) >> 5;
    if (e >= total) return;
    const int lane = threadIdx.x & 31;
    int s, d;
    if (e < Emain) { s = src[e]; d = dst[e]; } else { s = d = e - Emain; }
    const long od = (long)d * 128 + lane * 4;
    const long os = (long)s * 128 + lane * 4;
    const float4 a  = *(const float4*)(&ai[od]);
    const float4 bj = *(const float4*)(&aj[os]);
    const float4 bb = *(const float4*)(&battn[lane * 4]);
    const int4  me = *(const int4*)(&m_enc[od]);
    const float4 xs = *(const float4*)(&xlin[os]);
    const float w0 = __expf(lrelu(a.x + bj.x + bb.x) - dec_ord(me.x));
    const float w1 = __expf(lrelu(a.y + bj.y + bb.y) - dec_ord(me.y));
    const float w2 = __expf(lrelu(a.z + bj.z + bb.z) - dec_ord(me.z));
    const float w3 = __expf(lrelu(a.w + bj.w + bb.w) - dec_ord(me.w));
    float* dp = denom + od;
    float* np = num + od;
    atomicAdd(dp + 0, w0); atomicAdd(np + 0, w0 * xs.x);
    atomicAdd(dp + 1, w1); atomicAdd(np + 1, w1 * xs.y);
    atomicAdd(dp + 2, w2); atomicAdd(np + 2, w2 * xs.z);
    atomicAdd(dp + 3, w3); atomicAdd(np + 3, w3 * xs.w);
}

__global__ __launch_bounds__(256) void seg_norm_kernel(const float* __restrict__ num,
                                                       const float* __restrict__ denom,
                                                       float* __restrict__ out, long n) {
    for (long i = (long)blockIdx.x * blockDim.x + threadIdx.x; i < n;
         i += (long)gridDim.x * blockDim.x)
        out[i] = num[i] / (denom[i] + 1e-16f);
}

// ---------------------------------------------------------------------------
// Scoring head: wave per node -> logits + sigmoid, then per-batch softmax
// ---------------------------------------------------------------------------
__global__ __launch_bounds__(256) void score_kernel(
    const float* __restrict__ g1, const float* __restrict__ h,
    const float* __restrict__ Wp, const float* __restrict__ bp,
    const float* __restrict__ Ws, const float* __restrict__ bs,
    float* __restrict__ logits, float* __restrict__ sisr_out, int Nnodes, int NN)
{
    const int node = (blockIdx.x * 256 + threadIdx.x) >> 5;
    if (node >= Nnodes) return;
    const int lane = threadIdx.x & 31;
    const int b = node / NN;
    const float4 g  = *(const float4*)(&g1[(long)node * 128 + lane * 4]);
    const float4 hh = *(const float4*)(&h[b * 128 + lane * 4]);
    const float4 wp = *(const float4*)(&Wp[lane * 4]);
    const float4 ws = *(const float4*)(&Ws[lane * 4]);
    const float x0 = g.x * hh.x, x1 = g.y * hh.y, x2 = g.z * hh.z, x3 = g.w * hh.w;
    float p = x0 * wp.x + x1 * wp.y + x2 * wp.z + x3 * wp.w;
    float s = x0 * ws.x + x1 * ws.y + x2 * ws.z + x3 * ws.w;
#pragma unroll
    for (int o = 16; o > 0; o >>= 1) {
        p += __shfl_xor(p, o, 32);
        s += __shfl_xor(s, o, 32);
    }
    if (lane == 0) {
        logits[node]   = p + bp[0];
        sisr_out[node] = 1.f / (1.f + __expf(-(s + bs[0])));
    }
}

__global__ __launch_bounds__(256) void softmax_kernel(const float* __restrict__ logits,
                                                      float* __restrict__ prob, int NN) {
    const int b = blockIdx.x, tid = threadIdx.x;
    const int n1 = NN - 1;
    const float* row = logits + (long)b * NN + 1;   // skip node 0
    __shared__ float red[256];
    float mx = -3.4e38f;
    for (int i = tid; i < n1; i += 256) mx = fmaxf(mx, row[i]);
    red[tid] = mx; __syncthreads();
    for (int o = 128; o > 0; o >>= 1) {
        if (tid < o) red[tid] = fmaxf(red[tid], red[tid + o]);
        __syncthreads();
    }
    mx = red[0]; __syncthreads();
    float sum = 0.f;
    for (int i = tid; i < n1; i += 256) sum += __expf(row[i] - mx);
    red[tid] = sum; __syncthreads();
    for (int o = 128; o > 0; o >>= 1) {
        if (tid < o) red[tid] += red[tid + o];
        __syncthreads();
    }
    const float inv = 1.f / red[0];
    for (int i = tid; i < n1; i += 256)
        prob[(long)b * n1 + i] = __expf(row[i] - mx) * inv;
}

// ---------------------------------------------------------------------------
// Host-side orchestration
// ---------------------------------------------------------------------------
extern "C" void kernel_launch(void* const* d_in, const int* in_sizes, int n_in,
                              void* d_out, int out_size, void* d_ws, size_t ws_size,
                              hipStream_t stream) {
    const float* x        = (const float*)d_in[0];
    const int*   ei0      = (const int*)d_in[1];
    const int*   ei1      = (const int*)d_in[2];
    const float* state_   = (const float*)d_in[3];
    const float* input_   = (const float*)d_in[4];
    const float* W_in     = (const float*)d_in[5];
    const float* W_z      = (const float*)d_in[6];
    const float* W_r      = (const float*)d_in[7];
    const float* W_h      = (const float*)d_in[8];
    const float* g0_Wlin  = (const float*)d_in[9];
    const float* g0_blin  = (const float*)d_in[10];
    const float* g0_Wattn = (const float*)d_in[11];
    const float* g0_battn = (const float*)d_in[12];
    const float* g1_Wlin  = (const float*)d_in[13];
    const float* g1_blin  = (const float*)d_in[14];
    const float* g1_Wattn = (const float*)d_in[15];
    const float* g1_battn = (const float*)d_in[16];
    const float* Wp       = (const float*)d_in[17];
    const float* bp       = (const float*)d_in[18];
    const float* Ws       = (const float*)d_in[19];
    const float* bs       = (const float*)d_in[20];

    const int N  = in_sizes[0] / 128;          // 32000
    const int E  = in_sizes[1] / 2;            // 256000
    const int B  = in_sizes[3] / 128;          // 64
    const int NN = N / B;                      // 500
    const int L  = in_sizes[4] / (B * 128);    // 50
    const long NF = (long)N * 128;

    // workspace layout
    char* w = (char*)d_ws;
    float*    F0  = (float*)w;                 w += NF * 4;   // xlin
    float*    F1  = (float*)w;                 w += NF * 4;   // ai
    float*    F2  = (float*)w;                 w += NF * 4;   // aj
    float*    F3  = (float*)w;                 w += NF * 4;   // m_enc / layer output
    float*    F4  = (float*)w;                 w += NF * 4;   // denom
    float*    F5  = (float*)w;                 w += NF * 4;   // num
    _Float16* B0  = (_Float16*)w;              w += NF * 2;   // layer input f16
    _Float16* B1  = (_Float16*)w;              w += NF * 2;   // xlin f16
    _Float16* WB  = (_Float16*)w;              w += (128 * 128 + 128 * 256) * 2;
    float*    inp    = (float*)w;              w += (long)B * 128 * 4;
    float*    logits = (float*)w;              w += (long)B * NN * 4;

    float* out_prob = (float*)d_out;                       // B x (NN-1)
    float* out_sisr = out_prob + (long)B * (NN - 1);       // B x NN
    float* out_h    = out_sisr + (long)B * NN;             // B x 128

    const int total_e = E + N;
    const int eblk    = (total_e * 32 + 255) / 256;        // wave per edge
    const int gemm_blk = (N + 127) / 128;
    const int cvt_blk  = 2048;
    const int seg_blk  = 2048;
    const int node_blk = (N * 32 + 255) / 256;             // wave per node

    // --- GRU branch (independent) ---
    gru_inp_kernel<<<B, 128, 0, stream>>>(input_, W_in, inp, L);
    gru_gates_kernel<<<B, 128, 0, stream>>>(state_, inp, W_z, W_r, W_h, out_h);

    // --- GAT layer 0 ---
    cvt_f32_f16<<<cvt_blk, 256, 0, stream>>>(x, B0, (int)NF);
    cvt_f32_f16<<<64, 256, 0, stream>>>(g0_Wlin, WB, 128 * 128);
    cvt_f32_f16<<<128, 256, 0, stream>>>(g0_Wattn, WB + 128 * 128, 128 * 256);
    gemm_wmma_f16<<<gemm_blk, 256, 0, stream>>>(B0, WB, 128, g0_blin, F0, B1, N);
    gemm_wmma_f16<<<gemm_blk, 256, 0, stream>>>(B1, WB + 128 * 128, 256, nullptr, F1, nullptr, N);
    gemm_wmma_f16<<<gemm_blk, 256, 0, stream>>>(B1, WB + 128 * 128 + 128, 256, nullptr, F2, nullptr, N);
    seg_init_kernel<<<seg_blk, 256, 0, stream>>>((int*)F3, F4, F5, NF);
    edge_max_kernel<<<eblk, 256, 0, stream>>>(ei0, ei0 + E, E, total_e, F1, F2, g0_battn, (int*)F3);
    edge_sum_kernel<<<eblk, 256, 0, stream>>>(ei0, ei0 + E, E, total_e, F1, F2, g0_battn, F0,
                                              (const int*)F3, F4, F5);
    seg_norm_kernel<<<seg_blk, 256, 0, stream>>>(F5, F4, F3, NF);   // F3 = g0

    // --- GAT layer 1 ---
    cvt_f32_f16<<<cvt_blk, 256, 0, stream>>>(F3, B0, (int)NF);
    cvt_f32_f16<<<64, 256, 0, stream>>>(g1_Wlin, WB, 128 * 128);
    cvt_f32_f16<<<128, 256, 0, stream>>>(g1_Wattn, WB + 128 * 128, 128 * 256);
    gemm_wmma_f16<<<gemm_blk, 256, 0, stream>>>(B0, WB, 128, g1_blin, F0, B1, N);
    gemm_wmma_f16<<<gemm_blk, 256, 0, stream>>>(B1, WB + 128 * 128, 256, nullptr, F1, nullptr, N);
    gemm_wmma_f16<<<gemm_blk, 256, 0, stream>>>(B1, WB + 128 * 128 + 128, 256, nullptr, F2, nullptr, N);
    seg_init_kernel<<<seg_blk, 256, 0, stream>>>((int*)F3, F4, F5, NF);
    edge_max_kernel<<<eblk, 256, 0, stream>>>(ei1, ei1 + E, E, total_e, F1, F2, g1_battn, (int*)F3);
    edge_sum_kernel<<<eblk, 256, 0, stream>>>(ei1, ei1 + E, E, total_e, F1, F2, g1_battn, F0,
                                              (const int*)F3, F4, F5);
    seg_norm_kernel<<<seg_blk, 256, 0, stream>>>(F5, F4, F3, NF);   // F3 = g1

    // --- scoring head ---
    score_kernel<<<node_blk, 256, 0, stream>>>(F3, out_h, Wp, bp, Ws, bs, logits, out_sisr, N, NN);
    softmax_kernel<<<B, 256, 0, stream>>>(logits, out_prob, NN);
}
